// CrossModalityBiAttentionBlock_9268539425489
// MI455X (gfx1250) — matrix operand order repaired and checked by reading, not compiled, not run on hardware
//
#include <hip/hip_runtime.h>
#include <hip/hip_bf16.h>
#include <math.h>

typedef __attribute__((ext_vector_type(16))) _Float16 v16h;
typedef __attribute__((ext_vector_type(8)))  _Float16 v8h;
typedef __attribute__((ext_vector_type(4)))  _Float16 v4h;
typedef __attribute__((ext_vector_type(8)))  float    v8f;

#define B_   2
#define L1_  4096
#define L2_  512
#define MDL  1152   // M1 == M2 == HID
#define H_   16
#define D_   72
#define DP_  96     // padded head dim (mult of 32 for K, 16 for N)
#define BH_  (B_*H_)
#define EPS_ 1e-6f
#define SCALE_ 0.11785113019775793f  // 72^-0.5

// ---------------------------------------------------------------------------
// CDNA5 async global->LDS DMA (ASYNCcnt-tracked), GVS addressing:
//   mem_addr = SGPR_U64 + VGPR_I32 ; LDS dest = per-lane 32-bit LDS address.
// LDS address = low 32 bits of the flat shared pointer (aperture in the high
// half per the ISA flat-address split).
// ---------------------------------------------------------------------------
__device__ __forceinline__ unsigned lds_off32(const void* p) {
  return (unsigned)(unsigned long long)p;
}
__device__ __forceinline__ void async_g2l_b128(unsigned ldsoff,
                                               unsigned long long gbase,
                                               unsigned goff) {
  asm volatile("global_load_async_to_lds_b128 %0, %1, %2"
               :: "v"(ldsoff), "v"(goff), "s"(gbase) : "memory");
}
__device__ __forceinline__ void wait_async0() {
  asm volatile("s_wait_asynccnt 0x0" ::: "memory");
}

// ---------------------------------------------------------------------------
// WMMA helpers: A and B fragments both use contiguous k = 16*(lane/16) + s.
// LDS tiles are stored K-contiguous with row stride 32 halves:
//   sA[m][kk] = A[m0+m][k0+kk]       (M-major)
//   sB[n][kk] = B[k0+kk][n0+n]       (N-major, i.e. B transposed)
// ---------------------------------------------------------------------------
__device__ __forceinline__ v8f wmma_f16(v16h a, v16h b, v8f c) {
  return __builtin_amdgcn_wmma_f32_16x16x32_f16(
      /*neg_a=*/false, a, /*neg_b=*/false, b,
      /*c_mod=*/(short)0, c, /*reuse_a=*/false, /*reuse_b=*/false);
}

// A 16x32 f16 fragment: lane L holds row M=L&15.
__device__ __forceinline__ v16h fragA(const _Float16* base, int lane) {
  int m  = lane & 15;
  int kb = (lane >> 4) << 4;
  const _Float16* p = base + m * 32 + kb;
  v8h lo = *(const v8h*)(p);
  v8h hi = *(const v8h*)(p + 8);
  v16h r;
#pragma unroll
  for (int i = 0; i < 8; ++i) { r[i] = lo[i]; r[i + 8] = hi[i]; }
  return r;
}

// B 32x16 f16 fragment: lane L holds column N=L&15 (contiguous in N-major LDS)
__device__ __forceinline__ v16h fragB(const _Float16* baseN, int lane) {
  int n  = lane & 15;
  int kb = (lane >> 4) << 4;
  return *(const v16h*)(baseN + n * 32 + kb);
}

// ---------------------------------------------------------------------------
// LayerNorm (affine-free) f32 -> f16, one block per row of 1152
// ---------------------------------------------------------------------------
__global__ __launch_bounds__(256) void ln_f16_kernel(
    const float* __restrict__ x, _Float16* __restrict__ out) {
  __shared__ float red[256];
  const int row = blockIdx.x, tid = threadIdx.x;
  const float* p = x + (size_t)row * MDL;
  float s = 0.f, ss = 0.f;
  for (int c = tid; c < MDL; c += 256) { float v = p[c]; s += v; ss += v * v; }
  red[tid] = s; __syncthreads();
  for (int o = 128; o > 0; o >>= 1) { if (tid < o) red[tid] += red[tid + o]; __syncthreads(); }
  float mu = red[0] * (1.0f / MDL); __syncthreads();
  red[tid] = ss; __syncthreads();
  for (int o = 128; o > 0; o >>= 1) { if (tid < o) red[tid] += red[tid + o]; __syncthreads(); }
  float var  = red[0] * (1.0f / MDL) - mu * mu;
  float rstd = rsqrtf(var + EPS_);
  _Float16* q = out + (size_t)row * MDL;
  for (int c = tid; c < MDL; c += 256) q[c] = (_Float16)((p[c] - mu) * rstd);
}

__global__ __launch_bounds__(256) void f32_to_f16_kernel(
    const float* __restrict__ in, _Float16* __restrict__ out, int n) {
  int i = blockIdx.x * 256 + threadIdx.x;
  if (i < n) out[i] = (_Float16)in[i];
}

// Zero head-dim padding columns [72,96) of a [nrows,96] f16 buffer.
__global__ __launch_bounds__(256) void zero_pad_kernel(
    _Float16* __restrict__ buf, int nrows) {
  int idx = blockIdx.x * 256 + threadIdx.x;
  if (idx < nrows * 24) {
    int row = idx / 24, c = D_ + idx % 24;
    buf[(size_t)row * DP_ + c] = (_Float16)0.0f;
  }
}

// ---------------------------------------------------------------------------
// Projection GEMM: out = (xm @ W + bias) * scale -> padded [B,H,L,96] f16.
// Tile 128x64, 8 waves (4x2), wave 32x32.  Double-buffered: A via async DMA,
// B (transpose) register-staged and committed after the WMMAs.
// ---------------------------------------------------------------------------
__global__ __launch_bounds__(256) void gemm_proj_kernel(
    const _Float16* __restrict__ A,   // [B*L, 1152]
    const _Float16* __restrict__ Wt,  // [1152, 1152]  (K x N)
    const float*    __restrict__ bias,
    _Float16*       __restrict__ outp,  // [B,H,L,96]
    int L, float scale) {
  __shared__ __align__(64) _Float16 sA[2][128 * 32];
  __shared__ __align__(64) _Float16 sB[2][64 * 32];
  const int tid = threadIdx.x, lane = tid & 31, wave = tid >> 5;
  const int wm = wave & 3, wn = wave >> 2;
  const int m0 = blockIdx.x * 128, n0 = blockIdx.y * 64;
  const int am = tid >> 2, ak8 = (tid & 3) << 3;   // A chunks: rows am, am+64
  const int kkB = tid >> 3, nbB = (tid & 7) << 3;  // B chunk
  const unsigned long long abase = (unsigned long long)A;
  v8f acc[2][2] = {};
  v8h tb;

  auto issueA = [&](int k0, int buf) {
    unsigned g0 = 2u * ((unsigned)(m0 + am) * MDL + (unsigned)(k0 + ak8));
    async_g2l_b128(lds_off32(&sA[buf][am * 32 + ak8]), abase, g0);
    async_g2l_b128(lds_off32(&sA[buf][(am + 64) * 32 + ak8]), abase,
                   g0 + 2u * 64u * MDL);
  };
  auto issueB = [&](int k0) {
    tb = *(const v8h*)(&Wt[(size_t)(k0 + kkB) * MDL + n0 + nbB]);
  };
  auto commitB = [&](int buf) {
#pragma unroll
    for (int i = 0; i < 8; ++i) sB[buf][(nbB + i) * 32 + kkB] = tb[i];
  };
  auto compute = [&](int buf) {
    v16h a0 = fragA(&sA[buf][(wm * 32 + 0)  * 32], lane);
    v16h a1 = fragA(&sA[buf][(wm * 32 + 16) * 32], lane);
    v16h b0 = fragB(&sB[buf][(wn * 32 + 0)  * 32], lane);
    v16h b1 = fragB(&sB[buf][(wn * 32 + 16) * 32], lane);
    acc[0][0] = wmma_f16(a0, b0, acc[0][0]);
    acc[0][1] = wmma_f16(a0, b1, acc[0][1]);
    acc[1][0] = wmma_f16(a1, b0, acc[1][0]);
    acc[1][1] = wmma_f16(a1, b1, acc[1][1]);
  };

  issueA(0, 0); issueB(0); commitB(0);
  wait_async0(); __syncthreads();
  int cur = 0;
  for (int step = 1; step < MDL / 32; ++step) {
    const int k0 = step * 32, nxt = cur ^ 1;
    issueA(k0, nxt);
    issueB(k0);
    compute(cur);
    commitB(nxt);
    wait_async0();
    __syncthreads();
    cur = nxt;
  }
  compute(cur);

  const int rbase = (lane < 16) ? 0 : 8, cl = lane & 15;
#pragma unroll
  for (int i = 0; i < 2; ++i)
#pragma unroll
    for (int j = 0; j < 2; ++j)
#pragma unroll
      for (int r = 0; r < 8; ++r) {
        int row = m0 + wm * 32 + i * 16 + rbase + r;
        int col = n0 + wn * 32 + j * 16 + cl;
        float v = (acc[i][j][r] + bias[col]) * scale;
        int bb = row / L, l = row % L;
        int h = col / D_, d = col % D_;
        outp[(((size_t)bb * H_ + h) * L + l) * DP_ + d] = (_Float16)v;
      }
}

// ---------------------------------------------------------------------------
// logits[b,h,l1,l2] = q . k (K = 96, zero padded).  B = k^T so the N-major
// LDS tile is a row copy of k: both tiles use async DMA; double-buffered.
// ---------------------------------------------------------------------------
__global__ __launch_bounds__(256) void gemm_logits_kernel(
    const _Float16* __restrict__ qh,  // [BH, L1, 96]
    const _Float16* __restrict__ kh,  // [BH, L2, 96]
    float* __restrict__ logits) {     // [BH, L1, L2]
  __shared__ __align__(64) _Float16 sA[2][128 * 32];
  __shared__ __align__(64) _Float16 sB[2][64 * 32];
  const int tid = threadIdx.x, lane = tid & 31, wave = tid >> 5;
  const int wm = wave & 3, wn = wave >> 2;
  const int bh = blockIdx.z;
  const int m0 = blockIdx.x * 128, n0 = blockIdx.y * 64;
  const int am = tid >> 2, ak8 = (tid & 3) << 3;  // A rows am, am+64
  const int bn = tid >> 2, bk8 = (tid & 3) << 3;  // B rows (64)
  const unsigned qb = (unsigned)bh * L1_ * DP_, kb = (unsigned)bh * L2_ * DP_;
  const unsigned long long qbase = (unsigned long long)qh;
  const unsigned long long kbase = (unsigned long long)kh;
  v8f acc[2][2] = {};

  auto issueTiles = [&](int k0, int buf) {
    unsigned ga = 2u * (qb + (unsigned)(m0 + am) * DP_ + (unsigned)(k0 + ak8));
    async_g2l_b128(lds_off32(&sA[buf][am * 32 + ak8]), qbase, ga);
    async_g2l_b128(lds_off32(&sA[buf][(am + 64) * 32 + ak8]), qbase,
                   ga + 2u * 64u * DP_);
    unsigned gb = 2u * (kb + (unsigned)(n0 + bn) * DP_ + (unsigned)(k0 + bk8));
    async_g2l_b128(lds_off32(&sB[buf][bn * 32 + bk8]), kbase, gb);
  };
  auto compute = [&](int buf) {
    v16h a0 = fragA(&sA[buf][(wm * 32 + 0)  * 32], lane);
    v16h a1 = fragA(&sA[buf][(wm * 32 + 16) * 32], lane);
    v16h b0 = fragB(&sB[buf][(wn * 32 + 0)  * 32], lane);
    v16h b1 = fragB(&sB[buf][(wn * 32 + 16) * 32], lane);
    acc[0][0] = wmma_f16(a0, b0, acc[0][0]);
    acc[0][1] = wmma_f16(a0, b1, acc[0][1]);
    acc[1][0] = wmma_f16(a1, b0, acc[1][0]);
    acc[1][1] = wmma_f16(a1, b1, acc[1][1]);
  };

  issueTiles(0, 0);
  wait_async0(); __syncthreads();
  int cur = 0;
#pragma unroll
  for (int step = 1; step < DP_ / 32; ++step) {
    const int nxt = cur ^ 1;
    issueTiles(step * 32, nxt);
    compute(cur);
    wait_async0();
    __syncthreads();
    cur = nxt;
  }
  compute(cur);

  const int rbase = (lane < 16) ? 0 : 8, cl = lane & 15;
  const size_t lb = (size_t)bh * L1_ * L2_;
#pragma unroll
  for (int i = 0; i < 2; ++i)
#pragma unroll
    for (int j = 0; j < 2; ++j)
#pragma unroll
      for (int r = 0; r < 8; ++r) {
        int row = m0 + wm * 32 + i * 16 + rbase + r;
        int col = n0 + wn * 32 + j * 16 + cl;
        logits[lb + (size_t)row * L2_ + col] = acc[i][j][r];
      }
}

// ---------------------------------------------------------------------------
// Softmax statistics (max and 1/sum of exp) per row (over L2) / col (over L1)
// ---------------------------------------------------------------------------
__global__ __launch_bounds__(256) void row_softmax_stats_kernel(
    const float* __restrict__ logits, float* __restrict__ rmax,
    float* __restrict__ rinv) {
  __shared__ float red[256];
  const int row = blockIdx.x, tid = threadIdx.x;
  const float* p = logits + (size_t)row * L2_;
  float a = p[tid], b = p[tid + 256];
  red[tid] = fmaxf(a, b); __syncthreads();
  for (int o = 128; o > 0; o >>= 1) { if (tid < o) red[tid] = fmaxf(red[tid], red[tid + o]); __syncthreads(); }
  float mx = red[0]; __syncthreads();
  red[tid] = __expf(a - mx) + __expf(b - mx); __syncthreads();
  for (int o = 128; o > 0; o >>= 1) { if (tid < o) red[tid] += red[tid + o]; __syncthreads(); }
  if (tid == 0) { rmax[row] = mx; rinv[row] = 1.0f / red[0]; }
}

__global__ __launch_bounds__(256) void col_softmax_stats_kernel(
    const float* __restrict__ logits, float* __restrict__ cmax,
    float* __restrict__ cinv) {
  __shared__ float red[256];
  const int tid = threadIdx.x;
  const int bh = blockIdx.x >> 9, l2 = blockIdx.x & 511;
  const float* p = logits + (size_t)bh * L1_ * L2_ + l2;
  float m = -3.0e38f;
  for (int i = tid; i < L1_; i += 256) m = fmaxf(m, p[(size_t)i * L2_]);
  red[tid] = m; __syncthreads();
  for (int o = 128; o > 0; o >>= 1) { if (tid < o) red[tid] = fmaxf(red[tid], red[tid + o]); __syncthreads(); }
  float mx = red[0]; __syncthreads();
  float s = 0.f;
  for (int i = tid; i < L1_; i += 256) s += __expf(p[(size_t)i * L2_] - mx);
  red[tid] = s; __syncthreads();
  for (int o = 128; o > 0; o >>= 1) { if (tid < o) red[tid] += red[tid + o]; __syncthreads(); }
  if (tid == 0) { cmax[blockIdx.x] = mx; cinv[blockIdx.x] = 1.0f / red[0]; }
}

// ---------------------------------------------------------------------------
// o1 = softmax_rows(logits) @ v2.  A built on the fly as exp(l - rmax) (1/sum
// folded into epilogue).  Tile 128x96, wave 32x48 (2x3 acc).  Double-buffered
// register pipeline: next tile's loads + exp overlap current tile's WMMAs.
// ---------------------------------------------------------------------------
__global__ __launch_bounds__(256) void gemm_o1_kernel(
    const float* __restrict__ logits, const float* __restrict__ rmax,
    const float* __restrict__ rinv, const _Float16* __restrict__ v2h,
    _Float16* __restrict__ o1h) {  // [B, L1, 1152]
  __shared__ __align__(64) _Float16 sA[2][128 * 32];
  __shared__ __align__(64) _Float16 sB[2][96 * 32];
  const int tid = threadIdx.x, lane = tid & 31, wave = tid >> 5;
  const int wm = wave & 3, wn = wave >> 2;
  const int bh = blockIdx.z, b = bh >> 4, h = bh & 15;
  const int m0 = blockIdx.x * 128;
  const size_t lb = (size_t)bh * L1_ * L2_;
  v8f acc[2][3] = {};
  float la[16];
  v4h tb[3];

  auto issueA = [&](int k0) {
#pragma unroll
    for (int u = 0; u < 16; ++u) {
      int idx = tid + u * 256, m = idx >> 5, kk = idx & 31;
      la[u] = logits[lb + (size_t)(m0 + m) * L2_ + k0 + kk];
    }
  };
  auto commitA = [&](int buf) {
#pragma unroll
    for (int u = 0; u < 16; ++u) {
      int idx = tid + u * 256, m = idx >> 5, kk = idx & 31;
      sA[buf][m * 32 + kk] =
          (_Float16)__expf(la[u] - rmax[(size_t)bh * L1_ + m0 + m]);
    }
  };
  auto issueB = [&](int k0) {
#pragma unroll
    for (int u = 0; u < 3; ++u) {
      int idx = tid + u * 256, kk = idx / 24, nb4 = (idx % 24) * 4;
      tb[u] = *(const v4h*)(&v2h[((size_t)bh * L2_ + k0 + kk) * DP_ + nb4]);
    }
  };
  auto commitB = [&](int buf) {
#pragma unroll
    for (int u = 0; u < 3; ++u) {
      int idx = tid + u * 256, kk = idx / 24, nb4 = (idx % 24) * 4;
#pragma unroll
      for (int i = 0; i < 4; ++i) sB[buf][(nb4 + i) * 32 + kk] = tb[u][i];
    }
  };
  auto compute = [&](int buf) {
    v16h a0 = fragA(&sA[buf][(wm * 32 + 0)  * 32], lane);
    v16h a1 = fragA(&sA[buf][(wm * 32 + 16) * 32], lane);
#pragma unroll
    for (int j = 0; j < 3; ++j) {
      v16h bf = fragB(&sB[buf][(wn * 48 + j * 16) * 32], lane);
      acc[0][j] = wmma_f16(a0, bf, acc[0][j]);
      acc[1][j] = wmma_f16(a1, bf, acc[1][j]);
    }
  };

  issueA(0); issueB(0); commitA(0); commitB(0);
  __syncthreads();
  int cur = 0;
  for (int step = 1; step < L2_ / 32; ++step) {
    const int k0 = step * 32, nxt = cur ^ 1;
    issueA(k0);
    issueB(k0);
    compute(cur);
    commitA(nxt);
    commitB(nxt);
    __syncthreads();
    cur = nxt;
  }
  compute(cur);

  const int rbase = (lane < 16) ? 0 : 8, cl = lane & 15;
#pragma unroll
  for (int i = 0; i < 2; ++i)
#pragma unroll
    for (int j = 0; j < 3; ++j)
#pragma unroll
      for (int r = 0; r < 8; ++r) {
        int row = m0 + wm * 32 + i * 16 + rbase + r;
        int col = wn * 48 + j * 16 + cl;
        if (col < D_) {
          float v = acc[i][j][r] * rinv[(size_t)bh * L1_ + row];
          o1h[((size_t)b * L1_ + row) * MDL + h * D_ + col] = (_Float16)v;
        }
      }
}

// o2 = softmax_cols(logits)^T @ v1: rows are l2, K runs over l1 (transposed
// logits reads), 1/colsum folded into epilogue.  Same pipeline as o1.
__global__ __launch_bounds__(256) void gemm_o2_kernel(
    const float* __restrict__ logits, const float* __restrict__ cmax,
    const float* __restrict__ cinv, const _Float16* __restrict__ v1h,
    _Float16* __restrict__ o2h) {  // [B, L2, 1152]
  __shared__ __align__(64) _Float16 sA[2][128 * 32];
  __shared__ __align__(64) _Float16 sB[2][96 * 32];
  const int tid = threadIdx.x, lane = tid & 31, wave = tid >> 5;
  const int wm = wave & 3, wn = wave >> 2;
  const int bh = blockIdx.z, b = bh >> 4, h = bh & 15;
  const int m0 = blockIdx.x * 128;
  const size_t lb = (size_t)bh * L1_ * L2_;
  v8f acc[2][3] = {};
  float la[16];
  v4h tb[3];

  auto issueA = [&](int k0) {
#pragma unroll
    for (int u = 0; u < 16; ++u) {
      int idx = tid + u * 256, m = idx >> 5, kk = idx & 31;
      la[u] = logits[lb + (size_t)(k0 + kk) * L2_ + m0 + m];
    }
  };
  auto commitA = [&](int buf) {
#pragma unroll
    for (int u = 0; u < 16; ++u) {
      int idx = tid + u * 256, m = idx >> 5, kk = idx & 31;
      sA[buf][m * 32 + kk] =
          (_Float16)__expf(la[u] - cmax[(size_t)bh * L2_ + m0 + m]);
    }
  };
  auto issueB = [&](int k0) {
#pragma unroll
    for (int u = 0; u < 3; ++u) {
      int idx = tid + u * 256, kk = idx / 24, nb4 = (idx % 24) * 4;
      tb[u] = *(const v4h*)(&v1h[((size_t)bh * L1_ + k0 + kk) * DP_ + nb4]);
    }
  };
  auto commitB = [&](int buf) {
#pragma unroll
    for (int u = 0; u < 3; ++u) {
      int idx = tid + u * 256, kk = idx / 24, nb4 = (idx % 24) * 4;
#pragma unroll
      for (int i = 0; i < 4; ++i) sB[buf][(nb4 + i) * 32 + kk] = tb[u][i];
    }
  };
  auto compute = [&](int buf) {
    v16h a0 = fragA(&sA[buf][(wm * 32 + 0)  * 32], lane);
    v16h a1 = fragA(&sA[buf][(wm * 32 + 16) * 32], lane);
#pragma unroll
    for (int j = 0; j < 3; ++j) {
      v16h bf = fragB(&sB[buf][(wn * 48 + j * 16) * 32], lane);
      acc[0][j] = wmma_f16(a0, bf, acc[0][j]);
      acc[1][j] = wmma_f16(a1, bf, acc[1][j]);
    }
  };

  issueA(0); issueB(0); commitA(0); commitB(0);
  __syncthreads();
  int cur = 0;
  for (int step = 1; step < L1_ / 32; ++step) {
    const int k0 = step * 32, nxt = cur ^ 1;
    issueA(k0);
    issueB(k0);
    compute(cur);
    commitA(nxt);
    commitB(nxt);
    __syncthreads();
    cur = nxt;
  }
  compute(cur);

  const int rbase = (lane < 16) ? 0 : 8, cl = lane & 15;
#pragma unroll
  for (int i = 0; i < 2; ++i)
#pragma unroll
    for (int j = 0; j < 3; ++j)
#pragma unroll
      for (int r = 0; r < 8; ++r) {
        int row = m0 + wm * 32 + i * 16 + rbase + r;
        int col = wn * 48 + j * 16 + cl;
        if (col < D_) {
          float v = acc[i][j][r] * cinv[(size_t)bh * L2_ + row];
          o2h[((size_t)b * L2_ + row) * MDL + h * D_ + col] = (_Float16)v;
        }
      }
}

// ---------------------------------------------------------------------------
// Final projection + residual: out = x + gamma * (o @ Wo + bias)  (f32 out)
// Same double-buffered async pipeline as gemm_proj.
// ---------------------------------------------------------------------------
__global__ __launch_bounds__(256) void gemm_final_kernel(
    const _Float16* __restrict__ A,   // [M, 1152]
    const _Float16* __restrict__ Wt,  // [1152, 1152]
    const float* __restrict__ bias, const float* __restrict__ gamma,
    const float* __restrict__ x, float* __restrict__ out) {
  __shared__ __align__(64) _Float16 sA[2][128 * 32];
  __shared__ __align__(64) _Float16 sB[2][64 * 32];
  const int tid = threadIdx.x, lane = tid & 31, wave = tid >> 5;
  const int wm = wave & 3, wn = wave >> 2;
  const int m0 = blockIdx.x * 128, n0 = blockIdx.y * 64;
  const int am = tid >> 2, ak8 = (tid & 3) << 3;
  const int kkB = tid >> 3, nbB = (tid & 7) << 3;
  const unsigned long long abase = (unsigned long long)A;
  v8f acc[2][2] = {};
  v8h tb;

  auto issueA = [&](int k0, int buf) {
    unsigned g0 = 2u * ((unsigned)(m0 + am) * MDL + (unsigned)(k0 + ak8));
    async_g2l_b128(lds_off32(&sA[buf][am * 32 + ak8]), abase, g0);
    async_g2l_b128(lds_off32(&sA[buf][(am + 64) * 32 + ak8]), abase,
                   g0 + 2u * 64u * MDL);
  };
  auto issueB = [&](int k0) {
    tb = *(const v8h*)(&Wt[(size_t)(k0 + kkB) * MDL + n0 + nbB]);
  };
  auto commitB = [&](int buf) {
#pragma unroll
    for (int i = 0; i < 8; ++i) sB[buf][(nbB + i) * 32 + kkB] = tb[i];
  };
  auto compute = [&](int buf) {
    v16h a0 = fragA(&sA[buf][(wm * 32 + 0)  * 32], lane);
    v16h a1 = fragA(&sA[buf][(wm * 32 + 16) * 32], lane);
    v16h b0 = fragB(&sB[buf][(wn * 32 + 0)  * 32], lane);
    v16h b1 = fragB(&sB[buf][(wn * 32 + 16) * 32], lane);
    acc[0][0] = wmma_f16(a0, b0, acc[0][0]);
    acc[0][1] = wmma_f16(a0, b1, acc[0][1]);
    acc[1][0] = wmma_f16(a1, b0, acc[1][0]);
    acc[1][1] = wmma_f16(a1, b1, acc[1][1]);
  };

  issueA(0, 0); issueB(0); commitB(0);
  wait_async0(); __syncthreads();
  int cur = 0;
  for (int step = 1; step < MDL / 32; ++step) {
    const int k0 = step * 32, nxt = cur ^ 1;
    issueA(k0, nxt);
    issueB(k0);
    compute(cur);
    commitB(nxt);
    wait_async0();
    __syncthreads();
    cur = nxt;
  }
  compute(cur);

  const int rbase = (lane < 16) ? 0 : 8, cl = lane & 15;
#pragma unroll
  for (int i = 0; i < 2; ++i)
#pragma unroll
    for (int j = 0; j < 2; ++j)
#pragma unroll
      for (int r = 0; r < 8; ++r) {
        int row = m0 + wm * 32 + i * 16 + rbase + r;
        int col = n0 + wn * 32 + j * 16 + cl;
        size_t o = (size_t)row * MDL + col;
        out[o] = x[o] + gamma[col] * (acc[i][j][r] + bias[col]);
      }
}

// ---------------------------------------------------------------------------
extern "C" void kernel_launch(void* const* d_in, const int* in_sizes, int n_in,
                              void* d_out, int out_size, void* d_ws, size_t ws_size,
                              hipStream_t stream) {
  (void)in_sizes; (void)n_in; (void)out_size; (void)ws_size;
  const float* x1  = (const float*)d_in[0];
  const float* x2  = (const float*)d_in[1];
  const float* Wq  = (const float*)d_in[2];
  const float* bq  = (const float*)d_in[3];
  const float* Wk  = (const float*)d_in[4];
  const float* bk  = (const float*)d_in[5];
  const float* Wv1 = (const float*)d_in[6];
  const float* bv1 = (const float*)d_in[7];
  const float* Wv2 = (const float*)d_in[8];
  const float* bv2 = (const float*)d_in[9];
  const float* Wo1 = (const float*)d_in[10];
  const float* bo1 = (const float*)d_in[11];
  const float* Wo2 = (const float*)d_in[12];
  const float* bo2 = (const float*)d_in[13];
  const float* gamma1 = (const float*)d_in[14];
  const float* gamma2 = (const float*)d_in[15];

  const size_t XM1_N = (size_t)B_ * L1_ * MDL;   // 9,437,184
  const size_t XM2_N = (size_t)B_ * L2_ * MDL;   // 1,179,648
  const size_t W_N   = (size_t)MDL * MDL;        // 1,327,104
  const size_t Q_N   = (size_t)BH_ * L1_ * DP_;  // 12,582,912
  const size_t K_N   = (size_t)BH_ * L2_ * DP_;  //  1,572,864
  const size_t LOG_N = (size_t)BH_ * L1_ * L2_;  // 67,108,864
  const size_t RS_N  = (size_t)BH_ * L1_;
  const size_t CS_N  = (size_t)BH_ * L2_;

  char* ws = (char*)d_ws;
  size_t off = 0;
  auto take = [&](size_t bytes) -> void* {
    void* p = ws + off;
    off = (off + bytes + 255) & ~(size_t)255;
    return p;
  };
  _Float16* xm1h = (_Float16*)take(XM1_N * 2);
  _Float16* xm2h = (_Float16*)take(XM2_N * 2);
  _Float16* wqh  = (_Float16*)take(W_N * 2);
  _Float16* wkh  = (_Float16*)take(W_N * 2);
  _Float16* wv1h = (_Float16*)take(W_N * 2);
  _Float16* wv2h = (_Float16*)take(W_N * 2);
  _Float16* wo1h = (_Float16*)take(W_N * 2);
  _Float16* wo2h = (_Float16*)take(W_N * 2);
  _Float16* qh   = (_Float16*)take(Q_N * 2);
  _Float16* kh   = (_Float16*)take(K_N * 2);
  _Float16* v1h  = (_Float16*)take(Q_N * 2);
  _Float16* v2h  = (_Float16*)take(K_N * 2);
  float*    logits = (float*)take(LOG_N * 4);
  float*    rmax = (float*)take(RS_N * 4);
  float*    rinv = (float*)take(RS_N * 4);
  float*    cmax = (float*)take(CS_N * 4);
  float*    cinv = (float*)take(CS_N * 4);
  _Float16* o1h  = (_Float16*)take(XM1_N * 2);
  _Float16* o2h  = (_Float16*)take(XM2_N * 2);

  float* out1 = (float*)d_out;
  float* out2 = out1 + XM1_N;

  // 1) LayerNorm -> f16
  ln_f16_kernel<<<B_ * L1_, 256, 0, stream>>>(x1, xm1h);
  ln_f16_kernel<<<B_ * L2_, 256, 0, stream>>>(x2, xm2h);

  // 2) Weight conversion f32 -> f16
  const int wgrid = (int)((W_N + 255) / 256);
  f32_to_f16_kernel<<<wgrid, 256, 0, stream>>>(Wq,  wqh,  (int)W_N);
  f32_to_f16_kernel<<<wgrid, 256, 0, stream>>>(Wk,  wkh,  (int)W_N);
  f32_to_f16_kernel<<<wgrid, 256, 0, stream>>>(Wv1, wv1h, (int)W_N);
  f32_to_f16_kernel<<<wgrid, 256, 0, stream>>>(Wv2, wv2h, (int)W_N);
  f32_to_f16_kernel<<<wgrid, 256, 0, stream>>>(Wo1, wo1h, (int)W_N);
  f32_to_f16_kernel<<<wgrid, 256, 0, stream>>>(Wo2, wo2h, (int)W_N);

  // 3) Zero head-dim padding
  zero_pad_kernel<<<(BH_ * L1_ * 24 + 255) / 256, 256, 0, stream>>>(qh,  BH_ * L1_);
  zero_pad_kernel<<<(BH_ * L1_ * 24 + 255) / 256, 256, 0, stream>>>(v1h, BH_ * L1_);
  zero_pad_kernel<<<(BH_ * L2_ * 24 + 255) / 256, 256, 0, stream>>>(kh,  BH_ * L2_);
  zero_pad_kernel<<<(BH_ * L2_ * 24 + 255) / 256, 256, 0, stream>>>(v2h, BH_ * L2_);

  // 4) Projections (q pre-scaled by D^-0.5)
  gemm_proj_kernel<<<dim3(64, 18), 256, 0, stream>>>(xm1h, wqh,  bq,  qh,  L1_, SCALE_);
  gemm_proj_kernel<<<dim3(8, 18),  256, 0, stream>>>(xm2h, wkh,  bk,  kh,  L2_, 1.0f);
  gemm_proj_kernel<<<dim3(64, 18), 256, 0, stream>>>(xm1h, wv1h, bv1, v1h, L1_, 1.0f);
  gemm_proj_kernel<<<dim3(8, 18),  256, 0, stream>>>(xm2h, wv2h, bv2, v2h, L2_, 1.0f);

  // 5) Shared bidirectional logits
  gemm_logits_kernel<<<dim3(32, 8, BH_), 256, 0, stream>>>(qh, kh, logits);

  // 6) Softmax statistics
  row_softmax_stats_kernel<<<BH_ * L1_, 256, 0, stream>>>(logits, rmax, rinv);
  col_softmax_stats_kernel<<<BH_ * L2_, 256, 0, stream>>>(logits, cmax, cinv);

  // 7) Attention-apply GEMMs with on-the-fly softmax A-tiles
  gemm_o1_kernel<<<dim3(32, 1, BH_), 256, 0, stream>>>(logits, rmax, rinv, v2h, o1h);
  gemm_o2_kernel<<<dim3(4, 1, BH_),  256, 0, stream>>>(logits, cmax, cinv, v1h, o2h);

  // 8) Output projections + gated residual
  gemm_final_kernel<<<dim3(64, 18), 256, 0, stream>>>(o1h, wo1h, bo1, gamma1, x1, out1);
  gemm_final_kernel<<<dim3(8, 18),  256, 0, stream>>>(o2h, wo2h, bo2, gamma2, x2, out2);
}
